// VectorQuantiser_73959336837428
// MI455X (gfx1250) — compile-verified, matrix operand order checked by reading, not run on hardware
//
#include <hip/hip_runtime.h>

#define VQ_N    32768
#define VQ_F    768
#define VQ_C    64
#define VQ_P    8192
#define VQ_BETA 0.25f

#define LOSS_BLOCKS 256
#define CHUNK 256   // codebook pages staged in LDS per iteration (256*128B = 32KB)

typedef __attribute__((ext_vector_type(16))) __bf16 v16bf;
typedef __attribute__((ext_vector_type(8)))  __bf16 v8bf;
typedef __attribute__((ext_vector_type(8)))  float  v8f;

#if defined(__has_builtin)
#if __has_builtin(__builtin_amdgcn_global_load_async_to_lds_b128)
#define VQ_ASYNC_LDS 1
#endif
#endif

// Exact parameter types for the async-to-LDS builtin (from clang diagnostic):
// arg0: int4 in AS1 (global), arg1: int4 in AS3 (LDS), then imm offset, imm cpol.
typedef int vqint4 __attribute__((vector_size(16)));
typedef __attribute__((address_space(1))) vqint4 as1_int4;
typedef __attribute__((address_space(3))) vqint4 as3_int4;

#define WMMA_BF16(a, b, c) \
  __builtin_amdgcn_wmma_f32_16x16x32_bf16(false, (a), false, (b), (short)0, (c), false, false)

// Build a 16x32 (or 32x16) bf16 fragment from an f32 row-major row.
// Per ISA 7.12.2: lane L (half h=L/16) holds K = {8h..8h+7} in VGPR0-3 and
// K = {16+8h..16+8h+7} in VGPR4-7 -> v16bf elements [0..7] and [8..15].
static __device__ __forceinline__ v16bf frag_from_f32(const float* base, int h) {
  const float* p1 = base + 8 * h;
  const float* p2 = base + 16 + 8 * h;
  v16bf o;
#pragma unroll
  for (int i = 0; i < 8; ++i) { o[i] = (__bf16)p1[i]; o[i + 8] = (__bf16)p2[i]; }
  return o;
}

static __device__ __forceinline__ v16bf frag_from_bf16(const __bf16* base, int h) {
  v8bf lo = *(const v8bf*)(base + 8 * h);
  v8bf hi = *(const v8bf*)(base + 16 + 8 * h);
  v16bf o;
#pragma unroll
  for (int i = 0; i < 8; ++i) { o[i] = lo[i]; o[i + 8] = hi[i]; }
  return o;
}

// ---------------- K1: zq = normalise(z @ w_in^T + b_in), f32 + bf16 copies --
__global__ __launch_bounds__(256) void vq_project(
    const float* __restrict__ z, const float* __restrict__ w_in,
    const float* __restrict__ b_in,
    float* __restrict__ zq_f32, __bf16* __restrict__ zq_bf16) {
  const int lane = threadIdx.x & 31;
  const int wave = threadIdx.x >> 5;
  const int lr = lane & 15;
  const int h  = lane >> 4;
  const int m0 = (blockIdx.x * 8 + wave) * 16;

  const float* zrow = z + (size_t)(m0 + lr) * VQ_F;
  v8f acc[4];
  const v8f vzero = {};
#pragma unroll
  for (int t = 0; t < 4; ++t) acc[t] = vzero;

  for (int k0 = 0; k0 < VQ_F; k0 += 32) {
    v16bf a = frag_from_f32(zrow + k0, h);
#pragma unroll
    for (int t = 0; t < 4; ++t) {
      const float* wrow = w_in + (size_t)(t * 16 + lr) * VQ_F + k0;  // B[k][n] = w_in[n][k]
      v16bf b = frag_from_f32(wrow, h);
      acc[t] = WMMA_BF16(a, b, acc[t]);
    }
  }
  // bias: column n = 16t + lr, identical for all 8 rows a lane holds
#pragma unroll
  for (int t = 0; t < 4; ++t) {
    float bias = b_in[t * 16 + lr];
#pragma unroll
    for (int r = 0; r < 8; ++r) acc[t][r] += bias;
  }
  // row L2 norms: reduce over the 16 lanes (columns) of each half-wave
  float inv[8];
#pragma unroll
  for (int r = 0; r < 8; ++r) {
    float s = 0.f;
#pragma unroll
    for (int t = 0; t < 4; ++t) s += acc[t][r] * acc[t][r];
#pragma unroll
    for (int m = 1; m <= 8; m <<= 1) s += __shfl_xor(s, m, 32);
    inv[r] = rsqrtf(s);
  }
#pragma unroll
  for (int t = 0; t < 4; ++t)
#pragma unroll
    for (int r = 0; r < 8; ++r) {
      int row = m0 + r + 8 * h;
      int col = t * 16 + lr;
      float v = acc[t][r] * inv[r];
      zq_f32[(size_t)row * VQ_C + col] = v;
      zq_bf16[(size_t)row * VQ_C + col] = (__bf16)v;
    }
}

// ---------------- K2: cn = normalise(codebook), f32 + bf16 -----------------
__global__ __launch_bounds__(256) void vq_norm_codebook(
    const float* __restrict__ cb, float* __restrict__ cn_f32,
    __bf16* __restrict__ cn_bf16) {
  const int lane = threadIdx.x & 31;
  const int wave = threadIdx.x >> 5;
  const int p = blockIdx.x * 8 + wave;           // one wave per codebook row
  float2 v = ((const float2*)(cb + (size_t)p * VQ_C))[lane];
  float s = v.x * v.x + v.y * v.y;
#pragma unroll
  for (int m = 1; m <= 16; m <<= 1) s += __shfl_xor(s, m, 32);
  float inv = rsqrtf(s);
  float2 o; o.x = v.x * inv; o.y = v.y * inv;
  ((float2*)(cn_f32 + (size_t)p * VQ_C))[lane] = o;
  __bf16* br = cn_bf16 + (size_t)p * VQ_C + lane * 2;
  br[0] = (__bf16)o.x; br[1] = (__bf16)o.y;
}

// ---------------- K3: idx[n] = argmax_p zq[n].cn[p] (== argmin distance) ---
__global__ __launch_bounds__(256) void vq_argmax(
    const __bf16* __restrict__ zq_bf16, const __bf16* __restrict__ cn_bf16,
    int* __restrict__ idx_ws, int* __restrict__ idx_out) {
  __shared__ __bf16 lds_cn[CHUNK * VQ_C];  // 32 KB
  const int tid  = threadIdx.x;
  const int lane = tid & 31;
  const int wave = tid >> 5;
  const int lr = lane & 15;
  const int h  = lane >> 4;
  const int m0 = (blockIdx.x * 8 + wave) * 16;

  // A fragments (16 rows x K=64) resident for the whole search
  const __bf16* arow = zq_bf16 + (size_t)(m0 + lr) * VQ_C;
  v16bf a0 = frag_from_bf16(arow, h);
  v16bf a1 = frag_from_bf16(arow + 32, h);

  float best[8];
  int   bidx[8];
#pragma unroll
  for (int r = 0; r < 8; ++r) { best[r] = -1e30f; bidx[r] = 0; }

  for (int c0 = 0; c0 < VQ_P; c0 += CHUNK) {
    const __bf16* src = cn_bf16 + (size_t)c0 * VQ_C;
#ifdef VQ_ASYNC_LDS
#pragma unroll
    for (int j = 0; j < 8; ++j) {
      int e = tid + j * 256;  // 16-byte chunk id, 2048 total
      __builtin_amdgcn_global_load_async_to_lds_b128(
          (as1_int4*)(const void*)(src + (size_t)e * 8),
          (as3_int4*)(void*)(lds_cn + (size_t)e * 8), 0, 0);
    }
#if __has_builtin(__builtin_amdgcn_s_wait_asynccnt)
    __builtin_amdgcn_s_wait_asynccnt(0);
#else
    asm volatile("s_wait_asynccnt 0" ::: "memory");
#endif
#else
#pragma unroll
    for (int j = 0; j < 8; ++j) {
      int e = tid + j * 256;
      ((v8bf*)lds_cn)[e] = ((const v8bf*)src)[e];
    }
#endif
    __syncthreads();

    for (int pt = 0; pt < CHUNK; pt += 16) {
      const __bf16* brow = lds_cn + (pt + lr) * VQ_C;  // page column n = pt+lr
      v16bf b0 = frag_from_bf16(brow, h);
      v16bf b1 = frag_from_bf16(brow + 32, h);
      v8f c = {};
      c = WMMA_BF16(a0, b0, c);
      c = WMMA_BF16(a1, b1, c);
      const int page = c0 + pt + lr;
#pragma unroll
      for (int r = 0; r < 8; ++r)
        if (c[r] > best[r]) { best[r] = c[r]; bidx[r] = page; }  // strict > keeps first max
    }
    __syncthreads();
  }
  // argmax reduce across the 16 lanes (page residue classes) of each half
#pragma unroll
  for (int r = 0; r < 8; ++r) {
    float bv = best[r]; int bi = bidx[r];
#pragma unroll
    for (int m = 8; m >= 1; m >>= 1) {
      float ov = __shfl_xor(bv, m, 32);
      int   oi = __shfl_xor(bi, m, 32);
      if (ov > bv || (ov == bv && oi < bi)) { bv = ov; bi = oi; }
    }
    if (lr == 0) {
      int row = m0 + r + 8 * h;
      idx_ws[row]  = bi;
      idx_out[row] = bi;
    }
  }
}

// ---------------- K4/K5: loss = (1+beta)*mean((zq - cn[idx])^2) ------------
__global__ __launch_bounds__(256) void vq_loss_partial(
    const float* __restrict__ zq_f32, const float* __restrict__ cn_f32,
    const int* __restrict__ idx, float* __restrict__ partials) {
  __shared__ float red[256];
  const int tid = threadIdx.x;
  float s = 0.f;
  const size_t total = (size_t)VQ_N * VQ_C;
  for (size_t i = (size_t)blockIdx.x * 256 + tid; i < total;
       i += (size_t)LOSS_BLOCKS * 256) {
    int n = (int)(i >> 6), c = (int)(i & 63);
    float d = zq_f32[i] - cn_f32[(size_t)idx[n] * VQ_C + c];
    s += d * d;
  }
  red[tid] = s;
  __syncthreads();
  for (int w = 128; w > 0; w >>= 1) {
    if (tid < w) red[tid] += red[tid + w];
    __syncthreads();
  }
  if (tid == 0) partials[blockIdx.x] = red[0];
}

__global__ __launch_bounds__(256) void vq_loss_final(
    const float* __restrict__ partials, float* __restrict__ loss_out) {
  __shared__ float red[256];
  red[threadIdx.x] = partials[threadIdx.x];
  __syncthreads();
  for (int w = 128; w > 0; w >>= 1) {
    if (threadIdx.x < w) red[threadIdx.x] += red[threadIdx.x + w];
    __syncthreads();
  }
  if (threadIdx.x == 0)
    loss_out[0] = (1.0f + VQ_BETA) * red[0] / (float)((size_t)VQ_N * VQ_C);
}

// ---------------- K6: out = cn[idx] @ w_out^T + b_out ----------------------
__global__ __launch_bounds__(256) void vq_out_gemm(
    const __bf16* __restrict__ cn_bf16, const int* __restrict__ idx,
    const float* __restrict__ w_out, const float* __restrict__ b_out,
    float* __restrict__ out) {
  const int lane = threadIdx.x & 31;
  const int wave = threadIdx.x >> 5;
  const int lr = lane & 15;
  const int h  = lane >> 4;
  const int m0 = (blockIdx.x * 8 + wave) * 16;

  const __bf16* arow = cn_bf16 + (size_t)idx[m0 + lr] * VQ_C;  // gathered codes
  v16bf a0 = frag_from_bf16(arow, h);
  v16bf a1 = frag_from_bf16(arow + 32, h);

  for (int t = 0; t < VQ_F / 16; ++t) {
    const int f = t * 16 + lr;
    const float* wrow = w_out + (size_t)f * VQ_C;  // B[k][f] = w_out[f][k]
    v16bf b0 = frag_from_f32(wrow, h);
    v16bf b1 = frag_from_f32(wrow + 32, h);
    v8f c = {};
    c = WMMA_BF16(a0, b0, c);
    c = WMMA_BF16(a1, b1, c);
    float bias = b_out[f];
#pragma unroll
    for (int r = 0; r < 8; ++r)
      out[(size_t)(m0 + r + 8 * h) * VQ_F + f] = c[r] + bias;
  }
}

extern "C" void kernel_launch(void* const* d_in, const int* in_sizes, int n_in,
                              void* d_out, int out_size, void* d_ws, size_t ws_size,
                              hipStream_t stream) {
  const float* z     = (const float*)d_in[0];
  const float* w_in  = (const float*)d_in[1];
  const float* b_in  = (const float*)d_in[2];
  const float* w_out = (const float*)d_in[3];
  const float* b_out = (const float*)d_in[4];
  const float* cb    = (const float*)d_in[5];

  char* ws = (char*)d_ws;
  float*  zq_f32  = (float*)ws;   ws += (size_t)VQ_N * VQ_C * 4;
  __bf16* zq_bf16 = (__bf16*)ws;  ws += (size_t)VQ_N * VQ_C * 2;
  float*  cn_f32  = (float*)ws;   ws += (size_t)VQ_P * VQ_C * 4;
  __bf16* cn_bf16 = (__bf16*)ws;  ws += (size_t)VQ_P * VQ_C * 2;
  int*    idx_ws  = (int*)ws;     ws += (size_t)VQ_N * 4;
  float*  partials = (float*)ws;  ws += (size_t)LOSS_BLOCKS * 4;

  float* out      = (float*)d_out;
  float* loss_out = out + (size_t)VQ_N * VQ_F;
  int*   idx_out  = (int*)(loss_out + 1);

  vq_norm_codebook<<<dim3(VQ_P / 8), dim3(256), 0, stream>>>(cb, cn_f32, cn_bf16);
  vq_project<<<dim3(VQ_N / 128), dim3(256), 0, stream>>>(z, w_in, b_in, zq_f32, zq_bf16);
  vq_argmax<<<dim3(VQ_N / 128), dim3(256), 0, stream>>>(zq_bf16, cn_bf16, idx_ws, idx_out);
  vq_loss_partial<<<dim3(LOSS_BLOCKS), dim3(256), 0, stream>>>(zq_f32, cn_f32, idx_ws, partials);
  vq_loss_final<<<dim3(1), dim3(256), 0, stream>>>(partials, loss_out);
  vq_out_gemm<<<dim3(VQ_N / 128), dim3(256), 0, stream>>>(cn_bf16, idx_ws, w_out, b_out, out);
}
